// EmbeddingCoach_69088843924295
// MI455X (gfx1250) — matrix operand order repaired and checked by reading, compiled, not run on hardware
//
#include <hip/hip_runtime.h>

// ---------------------------------------------------------------------------
// Types for CDNA5 WMMA (wave32, gfx1250)
// ---------------------------------------------------------------------------
typedef __bf16 bf16_t;
typedef __bf16 v16bf __attribute__((ext_vector_type(16)));
typedef float  v8f   __attribute__((ext_vector_type(8)));
typedef unsigned int v4u __attribute__((ext_vector_type(4)));

__device__ __forceinline__ bf16_t f2bf(float f) {
  unsigned u = __builtin_bit_cast(unsigned, f);
  unsigned r = u + 0x7FFFu + ((u >> 16) & 1u);   // round-to-nearest-even
  unsigned short h = (unsigned short)(r >> 16);
  return __builtin_bit_cast(bf16_t, h);
}
__device__ __forceinline__ float lrelu(float x) { return x > 0.f ? x : 0.2f * x; }

// ---------------------------------------------------------------------------
// WMMA fragment loaders (bf16, 16x16x32).
// A (16xK tile, row-major): lane<16 holds row=lane, K [0..7] & [16..23];
// lane>=16 holds row=lane-16, K [8..15] & [24..31]  (ISA 7.12.2).
// B is stored transposed (N-major, "Bt"): per lane n=lane&15, same K split
// as A: lane<16 K [0..7] & [16..23]; lane>=16 K [8..15] & [24..31].
// ---------------------------------------------------------------------------
__device__ __forceinline__ v16bf load_fragA(const bf16_t* base, int ld, int lane) {
  const int r = lane & 15;
  const bf16_t* p = base + (size_t)r * ld + ((lane & 16) ? 8 : 0);
  union { v4u u[2]; v16bf v; } t;
  t.u[0] = *reinterpret_cast<const v4u*>(p);
  t.u[1] = *reinterpret_cast<const v4u*>(p + 16);
  return t.v;
}
__device__ __forceinline__ v16bf load_fragB(const bf16_t* base, int ld, int lane) {
  const int n = lane & 15;
  const bf16_t* p = base + (size_t)n * ld + ((lane & 16) ? 8 : 0);
  union { v4u u[2]; v16bf v; } t;
  t.u[0] = *reinterpret_cast<const v4u*>(p);
  t.u[1] = *reinterpret_cast<const v4u*>(p + 16);
  return t.v;
}

// ---------------------------------------------------------------------------
// Generic bf16 WMMA GEMM:  C(M x N) = A(M x K) @ Bt(N x K)^T  (+bias, act)
// Block = 128 threads = 4 waves; wave tile = 64(M) x 32(N) = 8 accumulators.
// grid = (N/32, M/256).
// rowmap: 0=identity, 1=ally->ent rows, 2=enemy->ent rows.
// ---------------------------------------------------------------------------
__global__ __launch_bounds__(128, 1) void gemm_bf16_wmma(
    const bf16_t* __restrict__ A, int lda,
    const bf16_t* __restrict__ Bt, int ldb, int K,
    const float* __restrict__ bias, int act,
    float* __restrict__ outF, int ldof,
    bf16_t* __restrict__ outB, int ldob, int rowmap)
{
  const int lane = threadIdx.x & 31;
  const int wv   = threadIdx.x >> 5;
  const int m0   = blockIdx.y * 256 + wv * 64;
  const int n0   = blockIdx.x * 32;

  v8f acc[4][2];
#pragma unroll
  for (int mi = 0; mi < 4; ++mi)
#pragma unroll
    for (int ni = 0; ni < 2; ++ni)
#pragma unroll
      for (int r = 0; r < 8; ++r) acc[mi][ni][r] = 0.f;

  for (int k0 = 0; k0 < K; k0 += 32) {
    v16bf a[4], b[2];
#pragma unroll
    for (int mi = 0; mi < 4; ++mi)
      a[mi] = load_fragA(A + (size_t)(m0 + mi * 16) * lda + k0, lda, lane);
#pragma unroll
    for (int ni = 0; ni < 2; ++ni)
      b[ni] = load_fragB(Bt + (size_t)(n0 + ni * 16) * ldb + k0, ldb, lane);
    if (k0 + 64 < K)   // emits global_prefetch_b8 for the A k-slab two steps ahead (k0+64)
      __builtin_prefetch(A + (size_t)(m0 + (lane & 15)) * lda + k0 + 64, 0, 1);
#pragma unroll
    for (int mi = 0; mi < 4; ++mi)
#pragma unroll
      for (int ni = 0; ni < 2; ++ni)
        acc[mi][ni] = __builtin_amdgcn_wmma_f32_16x16x32_bf16(
            false, a[mi], false, b[ni], (short)0, acc[mi][ni], false, false);
  }

  // Epilogue: C/D layout — VGPR r: lanes0-15 -> (M=r, N=lane); lanes16-31 -> (M=8+r).
  const int rbase = (lane & 16) ? 8 : 0;
  const int cl = lane & 15;
#pragma unroll
  for (int mi = 0; mi < 4; ++mi)
#pragma unroll
    for (int ni = 0; ni < 2; ++ni)
#pragma unroll
      for (int r = 0; r < 8; ++r) {
        int row = m0 + mi * 16 + rbase + r;
        int col = n0 + ni * 16 + cl;
        float v = acc[mi][ni][r];
        if (bias) v += bias[col];
        if (act) v = lrelu(v);
        int orow = row;
        if (rowmap == 1)      orow = ((row >> 6) << 7) + (row & 63);        // ally  -> ent
        else if (rowmap == 2) orow = ((row >> 6) << 7) + 64 + (row & 63);   // enemy -> ent
        if (outF) outF[(size_t)orow * ldof + col] = v;
        if (outB) outB[(size_t)orow * ldob + col] = f2bf(v);
      }
}

// ---------------------------------------------------------------------------
// Pack w2(64, 32*256)+b2 into Wc^T (256 x 2080 bf16): k<2048 -> w2[k>>5][ (k&31)*256+n ],
// k>=2048 -> b2[(k-2048)*256+n]  (bias folded into K-extension).
// ---------------------------------------------------------------------------
__global__ void pack_wc(const float* __restrict__ w2, const float* __restrict__ b2,
                        bf16_t* __restrict__ dst)
{
  int idx = blockIdx.x * blockDim.x + threadIdx.x;
  if (idx >= 256 * 2080) return;
  int n = idx / 2080, k = idx - n * 2080;
  float v = (k < 2048) ? w2[(size_t)(k >> 5) * 8192 + (size_t)(k & 31) * 256 + n]
                       : b2[(size_t)(k - 2048) * 256 + n];
  dst[(size_t)n * 2080 + k] = f2bf(v);
}

// Transpose-pack f32 (K x N) -> bf16 (N x K)
__global__ void pack_bt(const float* __restrict__ src, bf16_t* __restrict__ dst,
                        int N, int K)
{
  int idx = blockIdx.x * blockDim.x + threadIdx.x;
  if (idx >= N * K) return;
  int n = idx / K, k = idx - n * K;
  dst[idx] = f2bf(src[(size_t)k * N + n]);
}

// ---------------------------------------------------------------------------
// Per-entity: h1 = lrelu(s@w1+b1); P row = outer(h1, s) ++ s  (K = 2080, bf16)
// 1 entity per 64-thread block.
// ---------------------------------------------------------------------------
__global__ __launch_bounds__(64) void hyper_prep(
    const float* __restrict__ state, const float* __restrict__ w1,
    const float* __restrict__ b1, bf16_t* __restrict__ P)
{
  __shared__ float s[32];
  __shared__ float h1[64];
  const int e = blockIdx.x;
  const int t = threadIdx.x;
  if (t < 32) s[t] = state[(size_t)e * 32 + t];
  __syncthreads();
  float acc = b1[t];
#pragma unroll
  for (int a = 0; a < 32; ++a) acc += s[a] * w1[a * 64 + t];
  h1[t] = lrelu(acc);
  __syncthreads();
  bf16_t* row = P + (size_t)e * 2080;
  for (int k = t; k < 2080; k += 64) {
    float v = (k < 2048) ? h1[k >> 5] * s[k & 31] : s[k - 2048];
    row[k] = f2bf(v);
  }
}

// ---------------------------------------------------------------------------
// Linear attention + merge.  att has no softmax, so
//   out[i,h,f] = s_dst[i,h] * S1[h,f] + S2[h,f],
//   S1 = Σ_j ent,  S2 = Σ_j (s_src[j,h]+att_b) * ent.
// merged = Σ_h softmax(merge_w)[h] * out; keep i<64 (allies), emit bf16 for fc2.
// One block per batch element.
// ---------------------------------------------------------------------------
__global__ __launch_bounds__(256) void attn_merge(
    const float* __restrict__ ent, const float* __restrict__ att_w,
    const float* __restrict__ att_b, const float* __restrict__ merge_w,
    bf16_t* __restrict__ mergedB)
{
  __shared__ float ssrc[128 * 4], sdst[128 * 4];
  __shared__ float S1[4 * 64], S2[4 * 64];
  __shared__ float mw[4];
  const int b = blockIdx.x, t = threadIdx.x;
  if (t == 0) {
    float m = merge_w[0];
    for (int h = 1; h < 4; ++h) m = fmaxf(m, merge_w[h]);
    float sum = 0.f, e[4];
    for (int h = 0; h < 4; ++h) { e[h] = __expf(merge_w[h] - m); sum += e[h]; }
    for (int h = 0; h < 4; ++h) mw[h] = e[h] / sum;
  }
  // phase 1: per-(entity,head) attention scalars on lrelu(ent)
  for (int p = t; p < 512; p += 256) {
    int n = p >> 2, h = p & 3;
    const float* base = ent + ((size_t)(b * 128 + n)) * 256 + h * 64;
    float a1 = 0.f, a2 = 0.f;
    for (int f = 0; f < 64; ++f) {
      float v = lrelu(base[f]);
      a1 += v * att_w[f];
      a2 += v * att_w[64 + f];
    }
    ssrc[n * 4 + h] = a1;
    sdst[n * 4 + h] = a2;
  }
  __syncthreads();
  // phase 2: reductions over all 128 entities
  {
    int h = t >> 6, f = t & 63;
    float ab = att_b[0];
    float s1 = 0.f, s2 = 0.f;
    for (int n = 0; n < 128; ++n) {
      float v = ent[((size_t)(b * 128 + n)) * 256 + h * 64 + f];
      s1 += v;
      s2 += (ssrc[n * 4 + h] + ab) * v;
    }
    S1[h * 64 + f] = s1;
    S2[h * 64 + f] = s2;
  }
  __syncthreads();
  // phase 3: merged[i,f] for ally rows only (i < 64)
  for (int q = t; q < 4096; q += 256) {
    int i = q >> 6, f = q & 63;
    float m = 0.f;
#pragma unroll
    for (int h = 0; h < 4; ++h)
      m += mw[h] * (sdst[i * 4 + h] * S1[h * 64 + f] + S2[h * 64 + f]);
    mergedB[((size_t)(b * 64 + i)) * 64 + f] = f2bf(m);
  }
}

// ---------------------------------------------------------------------------
// Reparameterization: z = mu + eps*exp(0.5*clip(lv,-10,0)); outputs z|mu|lv.
// ---------------------------------------------------------------------------
__global__ void finalize(const float* __restrict__ dec, const float* __restrict__ eps,
                         float* __restrict__ out)
{
  const int TOT = 8192 * 256;
  int idx = blockIdx.x * blockDim.x + threadIdx.x;
  if (idx >= TOT) return;
  int r = idx >> 8, c = idx & 255;
  float mu = dec[(size_t)r * 512 + c];
  float lv = dec[(size_t)r * 512 + 256 + c];
  lv = fminf(fmaxf(lv, -10.f), 0.f);
  float z = mu + eps[idx] * __expf(0.5f * lv);
  out[idx] = z;
  out[TOT + idx] = mu;
  out[2 * TOT + idx] = lv;
}

// ---------------------------------------------------------------------------
extern "C" void kernel_launch(void* const* d_in, const int* in_sizes, int n_in,
                              void* d_out, int out_size, void* d_ws, size_t ws_size,
                              hipStream_t stream)
{
  (void)in_sizes; (void)n_in; (void)out_size; (void)ws_size;
  const float* ally_state  = (const float*)d_in[0];
  const float* enemy_state = (const float*)d_in[1];
  const float* eps         = (const float*)d_in[2];
  const float* ally_w1     = (const float*)d_in[3];
  const float* ally_b1     = (const float*)d_in[4];
  const float* ally_w2     = (const float*)d_in[5];
  const float* ally_b2     = (const float*)d_in[6];
  const float* enemy_w1    = (const float*)d_in[7];
  const float* enemy_b1    = (const float*)d_in[8];
  const float* enemy_w2    = (const float*)d_in[9];
  const float* enemy_b2    = (const float*)d_in[10];
  const float* att_w       = (const float*)d_in[11];
  const float* att_b       = (const float*)d_in[12];
  const float* merge_w     = (const float*)d_in[13];
  const float* fc2_w1      = (const float*)d_in[14];
  const float* fc2_b1      = (const float*)d_in[15];
  const float* fc2_w2      = (const float*)d_in[16];
  const float* fc2_b2      = (const float*)d_in[17];
  const float* dec_w1      = (const float*)d_in[18];
  const float* dec_b1      = (const float*)d_in[19];
  const float* dec_w2      = (const float*)d_in[20];
  const float* dec_b2      = (const float*)d_in[21];

  char* ws = (char*)d_ws;
  size_t off = 0;
  auto alloc = [&](size_t bytes) {
    void* p = ws + off;
    off = (off + bytes + 255) & ~(size_t)255;
    return p;
  };

  bf16_t* P       = (bf16_t*)alloc(8192ull * 2080 * 2);   // reused for ally then enemy
  bf16_t* WcA     = (bf16_t*)alloc(256ull * 2080 * 2);
  bf16_t* WcE     = (bf16_t*)alloc(256ull * 2080 * 2);
  bf16_t* fc2w1t  = (bf16_t*)alloc(256ull * 64 * 2);
  bf16_t* fc2w2t  = (bf16_t*)alloc(64ull * 256 * 2);
  bf16_t* decw1t  = (bf16_t*)alloc(256ull * 64 * 2);
  bf16_t* decw2t  = (bf16_t*)alloc(512ull * 256 * 2);
  float*  ent     = (float*) alloc(128ull * 128 * 256 * 4);
  bf16_t* mergedB = (bf16_t*)alloc(8192ull * 64 * 2);
  bf16_t* hid1B   = (bf16_t*)alloc(8192ull * 256 * 2);
  bf16_t* mixB    = (bf16_t*)alloc(8192ull * 64 * 2);
  bf16_t* dec1B   = (bf16_t*)alloc(8192ull * 256 * 2);
  float*  decF    = (float*) alloc(8192ull * 512 * 4);

  // Weight packing (bf16, B-transposed layouts)
  pack_wc<<<(256 * 2080 + 255) / 256, 256, 0, stream>>>(ally_w2, ally_b2, WcA);
  pack_wc<<<(256 * 2080 + 255) / 256, 256, 0, stream>>>(enemy_w2, enemy_b2, WcE);
  pack_bt<<<(256 * 64 + 255) / 256, 256, 0, stream>>>(fc2_w1, fc2w1t, 256, 64);
  pack_bt<<<(64 * 256 + 255) / 256, 256, 0, stream>>>(fc2_w2, fc2w2t, 64, 256);
  pack_bt<<<(256 * 64 + 255) / 256, 256, 0, stream>>>(dec_w1, decw1t, 256, 64);
  pack_bt<<<(512 * 256 + 255) / 256, 256, 0, stream>>>(dec_w2, decw2t, 512, 256);

  // Hypernetwork GEMMs (8192 x 256 x 2080, bf16 WMMA), writing into ent
  hyper_prep<<<8192, 64, 0, stream>>>(ally_state, ally_w1, ally_b1, P);
  gemm_bf16_wmma<<<dim3(8, 32), 128, 0, stream>>>(P, 2080, WcA, 2080, 2080,
      nullptr, 0, ent, 256, nullptr, 0, /*rowmap=*/1);
  hyper_prep<<<8192, 64, 0, stream>>>(enemy_state, enemy_w1, enemy_b1, P);
  gemm_bf16_wmma<<<dim3(8, 32), 128, 0, stream>>>(P, 2080, WcE, 2080, 2080,
      nullptr, 0, ent, 256, nullptr, 0, /*rowmap=*/2);

  // Linear attention + head merge -> merged (8192 x 64, bf16)
  attn_merge<<<128, 256, 0, stream>>>(ent, att_w, att_b, merge_w, mergedB);

  // fc2 + decoder GEMM chain (WMMA, fused bias/lrelu/bf16 epilogues)
  gemm_bf16_wmma<<<dim3(8, 32), 128, 0, stream>>>(mergedB, 64, fc2w1t, 64, 64,
      fc2_b1, 1, nullptr, 0, hid1B, 256, 0);
  gemm_bf16_wmma<<<dim3(2, 32), 128, 0, stream>>>(hid1B, 256, fc2w2t, 256, 256,
      fc2_b2, 0, nullptr, 0, mixB, 64, 0);
  gemm_bf16_wmma<<<dim3(8, 32), 128, 0, stream>>>(mixB, 64, decw1t, 64, 64,
      dec_b1, 1, nullptr, 0, dec1B, 256, 0);
  gemm_bf16_wmma<<<dim3(16, 32), 128, 0, stream>>>(dec1B, 256, decw2t, 256, 256,
      dec_b2, 0, decF, 512, nullptr, 0, 0);

  // Reparameterization + output packing (z, mu, log_var)
  finalize<<<(8192 * 256) / 256, 256, 0, stream>>>(decF, eps, (float*)d_out);
}
